// SRModel_56032143343870
// MI455X (gfx1250) — compile-verified, matrix operand order checked by reading, not run on hardware
//
#include <hip/hip_runtime.h>
#include <hip/hip_bf16.h>
#include <math.h>

typedef _Float16 f16;
typedef _Float16 v8h  __attribute__((ext_vector_type(8)));
typedef _Float16 v16h __attribute__((ext_vector_type(16)));
typedef float    v8f  __attribute__((ext_vector_type(8)));
typedef unsigned int u32x4 __attribute__((ext_vector_type(4)));
typedef int          i32x8 __attribute__((ext_vector_type(8)));
typedef int          i32x4 __attribute__((ext_vector_type(4)));

#if defined(__gfx1250__) && __has_builtin(__builtin_amdgcn_tensor_load_to_lds)
#define USE_TDM 1
#else
#define USE_TDM 0
#endif

#define DEV __device__ __forceinline__

DEV float sigmoidf_(float x) { return 1.0f / (1.0f + __expf(-x)); }
DEV float tanhf_(float x)    { float e = __expf(2.0f * x); return (e - 1.0f) / (e + 1.0f); }

// ---------------- weight conversion ----------------

// f32 -> f16 straight copy
__global__ void k_conv_f32_f16(const float* __restrict__ s, f16* __restrict__ d, int n) {
    int i = blockIdx.x * 256 + threadIdx.x;
    if (i < n) d[i] = (f16)s[i];
}

// src [K,N] f32 row-major  ->  dst [Npad,K] f16 (transposed, zero padded rows)
__global__ void k_tconv(const float* __restrict__ s, f16* __restrict__ d,
                        int K, int N, int Npad) {
    int i = blockIdx.x * 256 + threadIdx.x;
    if (i >= Npad * K) return;
    int n = i / K, k = i - n * K;
    float v = (n < N) ? s[(size_t)k * N + n] : 0.0f;
    d[(size_t)n * K + k] = (f16)v;
}

__global__ void k_bias_sum(const float* __restrict__ a, const float* __restrict__ b,
                           float* __restrict__ o, int n) {
    int i = blockIdx.x * 256 + threadIdx.x;
    if (i < n) o[i] = a[i] + b[i];
}

__global__ void k_zero_state(float* __restrict__ c, f16* __restrict__ h0,
                             f16* __restrict__ h1, int n) {
    int i = blockIdx.x * 256 + threadIdx.x;
    if (i < n) { c[i] = 0.0f; h0[i] = (f16)0.0f; h1[i] = (f16)0.0f; }
}

// ---------------- LayerNorm producers (emit f16 GEMM-A) ----------------

// layer-0: x (B,T,F=128) -> LN over F -> A16 rows ordered (t*B+b), width 128
__global__ void k_ln128(const float* __restrict__ x, const float* __restrict__ g,
                        const float* __restrict__ be, f16* __restrict__ out,
                        int Bsz, int T, int F) {
    int r = blockIdx.x;            // r = l*Bsz + b
    int l = r / Bsz, b = r - l * Bsz;
    int i = threadIdx.x;           // 0..127
    float v = x[((size_t)b * T + l) * F + i];
    __shared__ float s1[128], s2[128];
    s1[i] = v; s2[i] = v * v; __syncthreads();
    for (int st = 64; st > 0; st >>= 1) {
        if (i < st) { s1[i] += s1[i + st]; s2[i] += s2[i + st]; }
        __syncthreads();
    }
    float m  = s1[0] * (1.0f / 128.0f);
    float vr = s2[0] * (1.0f / 128.0f) - m * m;
    float rs = rsqrtf(vr + 1e-5f);
    out[(size_t)r * F + i] = (f16)((v - m) * rs * g[i] + be[i]);
}

// width-1024 LN: src rows (t*B+b); writes f16 A (optionally (B,L) transposed) and f32 copy
__global__ void k_ln1024(const float* __restrict__ x, const float* __restrict__ g,
                         const float* __restrict__ be, f16* __restrict__ out16,
                         float* __restrict__ outf, int transpose, int Bsz, int Lcur) {
    int r = blockIdx.x;
    int tid = threadIdx.x;
    const float* row = x + (size_t)r * 1024;
    float v[4]; float s = 0.0f, sq = 0.0f;
    for (int q = 0; q < 4; q++) { v[q] = row[tid + q * 256]; s += v[q]; sq += v[q] * v[q]; }
    __shared__ float s1[256], s2[256];
    s1[tid] = s; s2[tid] = sq; __syncthreads();
    for (int st = 128; st > 0; st >>= 1) {
        if (tid < st) { s1[tid] += s1[tid + st]; s2[tid] += s2[tid + st]; }
        __syncthreads();
    }
    float m  = s1[0] * (1.0f / 1024.0f);
    float vr = s2[0] * (1.0f / 1024.0f) - m * m;
    float rs = rsqrtf(vr + 1e-5f);
    size_t orow = r;
    if (transpose) { int l = r / Bsz, b = r - l * Bsz; orow = (size_t)b * Lcur + l; }
    for (int q = 0; q < 4; q++) {
        int i = tid + q * 256;
        float y = (v[q] - m) * rs * g[i] + be[i];
        out16[orow * 1024 + i] = (f16)y;
        if (outf) outf[(size_t)r * 1024 + i] = y;
    }
}

// ---------------- WMMA GEMM:  C[M,Nout] = A[M,K](f16) * BT[N,K](f16)^T (+bias) ----------------

__global__ __launch_bounds__(256)
void k_gemm_nt(const f16* __restrict__ A, const f16* __restrict__ BT,
               float* __restrict__ C, const float* __restrict__ bias,
               int M, int N, int K, int Nout) {
    __shared__ f16 sA[64 * 32];
    __shared__ f16 sB[128 * 32];
    int tid  = threadIdx.x;
    int m0   = blockIdx.y * 64;
    int n0   = blockIdx.x * 128;
    int wid  = tid >> 5, lane = tid & 31;
    int mw   = (wid >> 1) * 16;      // wave row offset: 0/16/32/48
    int nw   = (wid & 1) * 64;       // wave col offset: 0/64
    int lrow = lane & 15;
    int kbase = (lane >= 16) ? 8 : 0;

    v8f zf = {0.f, 0.f, 0.f, 0.f, 0.f, 0.f, 0.f, 0.f};
    v8f acc[4]; acc[0] = zf; acc[1] = zf; acc[2] = zf; acc[3] = zf;

    int arow = tid >> 2; int ak = (tid & 3) * 8;     // A tile: 64x32
    int brow = tid >> 1; int bk = (tid & 1) * 16;    // B tile: 128x32

    for (int k0 = 0; k0 < K; k0 += 32) {
        *(v8h*)&sA[arow * 32 + ak] = *(const v8h*)&A[(size_t)(m0 + arow) * K + k0 + ak];
        *(v8h*)&sB[brow * 32 + bk]     = *(const v8h*)&BT[(size_t)(n0 + brow) * K + k0 + bk];
        *(v8h*)&sB[brow * 32 + bk + 8] = *(const v8h*)&BT[(size_t)(n0 + brow) * K + k0 + bk + 8];
        __syncthreads();

        v16h af;
        {
            v8h lo = *(const v8h*)&sA[(mw + lrow) * 32 + kbase];
            v8h hi = *(const v8h*)&sA[(mw + lrow) * 32 + 16 + kbase];
            for (int e = 0; e < 8; e++) { af[e] = lo[e]; af[e + 8] = hi[e]; }
        }
        for (int s2 = 0; s2 < 4; s2++) {
            int col = nw + s2 * 16 + lrow;
            v16h bf;
            v8h lo = *(const v8h*)&sB[col * 32 + kbase];
            v8h hi = *(const v8h*)&sB[col * 32 + 16 + kbase];
            for (int e = 0; e < 8; e++) { bf[e] = lo[e]; bf[e + 8] = hi[e]; }
            acc[s2] = __builtin_amdgcn_wmma_f32_16x16x32_f16(
                false, af, false, bf, (short)0, acc[s2], false, false);
        }
        __syncthreads();
    }

    int rbase = (lane >= 16) ? 8 : 0;
    for (int s2 = 0; s2 < 4; s2++) {
        int col = n0 + nw + s2 * 16 + lrow;
        if (col < Nout) {
            float bv = bias ? bias[col] : 0.0f;
            for (int e = 0; e < 8; e++) {
                int row = m0 + mw + e + rbase;
                C[(size_t)row * Nout + col] = acc[s2][e] + bv;
            }
        }
    }
}

// ---------------- SRU recurrence scan (per (b,j) thread, loop over L) ----------------

__global__ void k_sru_scan(const float* __restrict__ U, const float* __restrict__ res,
                           const float* __restrict__ vc, const float* __restrict__ bias,
                           float* __restrict__ hs, int L, int Bsz, int kk) {
    int idx = blockIdx.x * 256 + threadIdx.x;   // b*1024 + j
    int b = idx >> 10, j = idx & 1023;
    float vf = vc[j], vr = vc[1024 + j];
    float bf = bias[j], br = bias[1024 + j];
    float c = 0.0f;
    size_t tstride = (size_t)Bsz * kk * 1024;
    for (int t = 0; t < L; t++) {
        size_t rowi = (size_t)(t * Bsz + b);
        size_t base = rowi * kk * 1024 + j;
        __builtin_prefetch(&U[base + tstride], 0, 1);   // global_prefetch for next step
        float u0 = U[base], u1 = U[base + 1024], u2 = U[base + 2048];
        float rt = (kk == 4) ? U[base + 3072] : res[rowi * 1024 + j];
        float f = sigmoidf_(u1 + vf * c + bf);
        float r = sigmoidf_(u2 + vr * c + br);
        c = f * c + (1.0f - f) * u0;
        hs[rowi * 1024 + j] = r * c + (1.0f - r) * rt;
    }
}

// ---------------- temporal mean-pool pairs ----------------

__global__ void k_pool(const float* __restrict__ in, float* __restrict__ out, int n) {
    int i = blockIdx.x * 256 + threadIdx.x;
    if (i >= n) return;
    int row = i >> 10, j = i & 1023;
    int l = row >> 4, b = row & 15;
    size_t r0 = ((size_t)(2 * l) * 16 + b) * 1024 + j;
    size_t r1 = ((size_t)(2 * l + 1) * 16 + b) * 1024 + j;
    out[i] = 0.5f * (in[r0] + in[r1]);
}

// ---------------- LSTM step: gates = pre_t + h @ Whh^T, then cell update ----------------
// grid: 16 blocks, each owns 64 hidden columns; 8 waves x 2 WMMA tiles = 16 gate tiles.
// h (16x1024 f16 = 32KB, contiguous) is staged into LDS with the Tensor Data Mover
// when available: one 1-D tile descriptor (data_size=8B, 4096 units), issued by wave 0,
// completion via s_wait_tensorcnt + workgroup barrier.

__global__ __launch_bounds__(256)
void k_lstm_step(const float* __restrict__ pre_t, const f16* __restrict__ Whh16,
                 const f16* __restrict__ hin, f16* __restrict__ hout,
                 float* __restrict__ cbuf, float* __restrict__ hsout) {
    __shared__ f16 sH[16 * 1024];          // whole h (16 batch x 1024) : 32 KB
    __shared__ float sG[16 * 16 * 16];     // 16 gate tiles of 16x16    : 16 KB
    int tid = threadIdx.x;

#if USE_TDM
    if (tid < 32) {                        // wave 0 issues the TDM transfer
        unsigned long long ga = (unsigned long long)(uintptr_t)hin;
        unsigned int lds = (unsigned int)(uintptr_t)&sH[0];   // low 32 bits = LDS offset
        u32x4 g0;
        g0[0] = 1u;                                            // count=1, is_restore=0
        g0[1] = lds;                                           // lds_addr
        g0[2] = (unsigned int)(ga & 0xFFFFFFFFu);              // global_addr[31:0] (in [95:64])
        g0[3] = (unsigned int)((ga >> 32) & 0x01FFFFFFu)       // global_addr[56:32]
              | 0x80000000u;                                   // type = 2 ("image")
        i32x8 g1;
        g1[0] = (int)(3u << 16);          // workgroup_mask=0, data_size=3 (8 bytes)
        g1[1] = (int)(4096u << 16);       // tensor_dim0[15:0]=4096 (8B units)
        g1[2] = (int)(1u << 16);          // tensor_dim0[31:16]=0, tensor_dim1=1
        g1[3] = (int)(4096u << 16);       // tile_dim0 = 4096
        g1[4] = 0;                        // tile_dim1 = 0 (unused), tile_dim2 = 0
        g1[5] = 4096;                     // tensor_dim0_stride = 4096
        g1[6] = 0;
        g1[7] = 0;
        i32x4 gz4 = {0, 0, 0, 0};                     // D# groups 2/3 (disabled, <=2D)
        i32x8 gz8 = {0, 0, 0, 0, 0, 0, 0, 0};         // trailing group (clang-23 form)
        __builtin_amdgcn_tensor_load_to_lds(g0, g1, gz4, gz4, gz8, 0);
        __builtin_amdgcn_s_wait_tensorcnt(0);
    }
#else
    for (int q = 0; q < 8; q++) {
        int off = tid * 64 + q * 8;
        *(v8h*)&sH[off] = *(const v8h*)&hin[off];
    }
#endif
    __syncthreads();

    int wid = tid >> 5, lane = tid & 31;
    int lrow = lane & 15;
    int kbase = (lane >= 16) ? 8 : 0;
    int hb = blockIdx.x;                   // hidden block of 64 cols
    int tt0 = wid * 2;

    v8f zf = {0.f, 0.f, 0.f, 0.f, 0.f, 0.f, 0.f, 0.f};
    v8f acc[2]; acc[0] = zf; acc[1] = zf;

    for (int k0 = 0; k0 < 1024; k0 += 32) {
        v16h af;
        {
            v8h lo = *(const v8h*)&sH[lrow * 1024 + k0 + kbase];
            v8h hi = *(const v8h*)&sH[lrow * 1024 + k0 + 16 + kbase];
            for (int e = 0; e < 8; e++) { af[e] = lo[e]; af[e + 8] = hi[e]; }
        }
        for (int u = 0; u < 2; u++) {
            int tt = tt0 + u;
            int g = tt >> 2, s2 = tt & 3;
            int col = g * 1024 + hb * 64 + s2 * 16 + lrow;
            v16h bf;
            const f16* bp = Whh16 + (size_t)col * 1024 + k0 + kbase;
            v8h lo = *(const v8h*)bp;
            v8h hi = *(const v8h*)(bp + 16);
            for (int e = 0; e < 8; e++) { bf[e] = lo[e]; bf[e + 8] = hi[e]; }
            acc[u] = __builtin_amdgcn_wmma_f32_16x16x32_f16(
                false, af, false, bf, (short)0, acc[u], false, false);
        }
    }

    int rbase = (lane >= 16) ? 8 : 0;
    for (int u = 0; u < 2; u++) {
        int tt = tt0 + u;
        for (int e = 0; e < 8; e++)
            sG[(tt * 16 + e + rbase) * 16 + lrow] = acc[u][e];
    }
    __syncthreads();

    for (int it = tid; it < 1024; it += 256) {
        int b = it >> 6, jj = it & 63;
        int s2 = jj >> 4, cw = jj & 15;
        int j = hb * 64 + jj;
        float gi = sG[((0 * 4 + s2) * 16 + b) * 16 + cw] + pre_t[(size_t)b * 4096 + j];
        float gf = sG[((1 * 4 + s2) * 16 + b) * 16 + cw] + pre_t[(size_t)b * 4096 + 1024 + j];
        float gg = sG[((2 * 4 + s2) * 16 + b) * 16 + cw] + pre_t[(size_t)b * 4096 + 2048 + j];
        float go = sG[((3 * 4 + s2) * 16 + b) * 16 + cw] + pre_t[(size_t)b * 4096 + 3072 + j];
        float cp = cbuf[b * 1024 + j];
        float c  = sigmoidf_(gf) * cp + sigmoidf_(gi) * tanhf_(gg);
        float hn = sigmoidf_(go) * tanhf_(c);
        cbuf[b * 1024 + j] = c;
        hout[b * 1024 + j] = (f16)hn;
        hsout[(size_t)b * 1024 + j] = hn;
    }
}

// ---------------- host orchestration ----------------

extern "C" void kernel_launch(void* const* d_in, const int* in_sizes, int n_in,
                              void* d_out, int out_size, void* d_ws, size_t ws_size,
                              hipStream_t stream) {
    (void)in_sizes; (void)n_in; (void)out_size; (void)ws_size;
    const int B = 16, T = 1024, F = 128, H = 1024;
    const int L2 = T / 2;                 // 512 after pooling
    const int Nout = 1001;

    const float* x    = (const float*)d_in[0];
    const float* w0   = (const float*)d_in[1];
    const float* vc0  = (const float*)d_in[2];
    const float* b0   = (const float*)d_in[3];
    const float* g0   = (const float*)d_in[4];
    const float* be0  = (const float*)d_in[5];
    const float* Ws   = (const float*)d_in[6];
    const float* VCs  = (const float*)d_in[7];
    const float* Bs   = (const float*)d_in[8];
    const float* Gs   = (const float*)d_in[9];
    const float* Bes  = (const float*)d_in[10];
    const float* Wih  = (const float*)d_in[11];
    const float* Whh  = (const float*)d_in[12];
    const float* bih  = (const float*)d_in[13];
    const float* bhh  = (const float*)d_in[14];
    const float* gh   = (const float*)d_in[15];
    const float* bh   = (const float*)d_in[16];
    const float* Wout = (const float*)d_in[17];
    float* outp = (float*)d_out;

    // workspace carve-up
    char* ws = (char*)d_ws;
    size_t off = 0;
    auto carve = [&](size_t bytes) { size_t o = off; off += (bytes + 255) & ~(size_t)255; return o; };
    f16*   w0T    = (f16*)(ws + carve((size_t)4096 * 128 * 2));
    f16*   WsT    = (f16*)(ws + carve((size_t)6 * 3072 * 1024 * 2));
    f16*   Wih16  = (f16*)(ws + carve((size_t)4096 * 1024 * 2));
    f16*   Whh16  = (f16*)(ws + carve((size_t)4096 * 1024 * 2));
    f16*   WoutT  = (f16*)(ws + carve((size_t)1024 * 1024 * 2));
    float* biasL  = (float*)(ws + carve((size_t)4096 * 4));
    f16*   A16    = (f16*)(ws + carve((size_t)16384 * 1024 * 2));
    float* resb   = (float*)(ws + carve((size_t)16384 * 1024 * 4));
    float* Ub     = (float*)(ws + carve((size_t)16384 * 4096 * 4));
    float* hsb    = (float*)(ws + carve((size_t)16384 * 1024 * 4));
    float* hs2    = (float*)(ws + carve((size_t)8192 * 1024 * 4));
    float* cbuf   = (float*)(ws + carve((size_t)16384 * 4));
    f16*   hb16   = (f16*)(ws + carve((size_t)2 * 16384 * 2));

    dim3 blk(256);
    // weight conversions to f16 [N,K]
    k_tconv<<<(4096 * 128 + 255) / 256, blk, 0, stream>>>(w0, w0T, 128, 4096, 4096);
    for (int i = 0; i < 6; i++)
        k_tconv<<<(3072 * 1024 + 255) / 256, blk, 0, stream>>>(
            Ws + (size_t)i * 1024 * 3072, WsT + (size_t)i * 3072 * 1024, 1024, 3072, 3072);
    k_conv_f32_f16<<<(4096 * 1024 + 255) / 256, blk, 0, stream>>>(Wih, Wih16, 4096 * 1024);
    k_conv_f32_f16<<<(4096 * 1024 + 255) / 256, blk, 0, stream>>>(Whh, Whh16, 4096 * 1024);
    k_tconv<<<(1024 * 1024 + 255) / 256, blk, 0, stream>>>(Wout, WoutT, 1024, Nout, 1024);
    k_bias_sum<<<16, blk, 0, stream>>>(bih, bhh, biasL, 4096);

    // ---- SRU layer 0 (k = 4, K = 128) ----
    k_ln128<<<T * B, dim3(128), 0, stream>>>(x, g0, be0, A16, B, T, F);
    k_gemm_nt<<<dim3(4096 / 128, 16384 / 64), blk, 0, stream>>>(
        A16, w0T, Ub, nullptr, 16384, 4096, 128, 4096);
    k_sru_scan<<<64, blk, 0, stream>>>(Ub, nullptr, vc0, b0, hsb, T, B, 4);

    // ---- SRU layers 1..3 (k = 3), L = 1024 ----
    for (int i = 0; i < 3; i++) {
        k_ln1024<<<16384, blk, 0, stream>>>(hsb, Gs + i * H, Bes + i * H, A16, resb, 0, B, T);
        k_gemm_nt<<<dim3(3072 / 128, 16384 / 64), blk, 0, stream>>>(
            A16, WsT + (size_t)i * 3072 * 1024, Ub, nullptr, 16384, 3072, 1024, 3072);
        k_sru_scan<<<64, blk, 0, stream>>>(Ub, resb, VCs + i * 2 * H, Bs + i * 2 * H, hsb, T, B, 3);
    }

    // ---- pool pairs over time: L 1024 -> 512 ----
    k_pool<<<(L2 * B * H + 255) / 256, blk, 0, stream>>>(hsb, hs2, L2 * B * H);

    // ---- SRU layers 4..6 (k = 3), L = 512 ----
    for (int i = 3; i < 6; i++) {
        k_ln1024<<<8192, blk, 0, stream>>>(hs2, Gs + i * H, Bes + i * H, A16, resb, 0, B, L2);
        k_gemm_nt<<<dim3(3072 / 128, 8192 / 64), blk, 0, stream>>>(
            A16, WsT + (size_t)i * 3072 * 1024, Ub, nullptr, 8192, 3072, 1024, 3072);
        k_sru_scan<<<64, blk, 0, stream>>>(Ub, resb, VCs + i * 2 * H, Bs + i * 2 * H, hs2, L2, B, 3);
    }

    // ---- LSTM input projection: pre = x @ Wih^T + (bih+bhh) ----
    k_conv_f32_f16<<<(8192 * 1024 + 255) / 256, blk, 0, stream>>>(hs2, A16, 8192 * 1024);
    k_gemm_nt<<<dim3(4096 / 128, 8192 / 64), blk, 0, stream>>>(
        A16, Wih16, Ub, biasL, 8192, 4096, 1024, 4096);

    // ---- LSTM recurrence: 512 sequential steps, WMMA + TDM per step ----
    k_zero_state<<<64, blk, 0, stream>>>(cbuf, hb16, hb16 + 16384, 16384);
    for (int t = 0; t < L2; t++) {
        const f16* hin  = hb16 + (size_t)(t & 1) * 16384;
        f16*       hout = hb16 + (size_t)((t + 1) & 1) * 16384;
        k_lstm_step<<<16, blk, 0, stream>>>(
            Ub + (size_t)t * 16 * 4096, Whh16, hin, hout,
            cbuf, hsb + (size_t)t * 16 * 1024);
    }

    // ---- final LN (with (L,B)->(B,L) transpose) + output GEMM ----
    k_ln1024<<<8192, blk, 0, stream>>>(hsb, gh, bh, A16, nullptr, 1, B, L2);
    k_gemm_nt<<<dim3(1024 / 128, 8192 / 64), blk, 0, stream>>>(
        A16, WoutT, outp, nullptr, 8192, 1024, 1024, Nout);
}
